// GCONVCheb_67448166416868
// MI455X (gfx1250) — matrix operand order repaired and checked by reading, not compiled, not run on hardware
//
#include <hip/hip_runtime.h>
#include <hip/hip_bf16.h>
#include <math.h>

// ---------------------------------------------------------------------------
// GConvGRU (ChebConv, K=5, H0=0) x3  +  per-edge MLP + softmax  for gfx1250.
// Matrix work (Tx_k @ W, dense MLP) runs on v_wmma_f32_16x16x32_f16 (wave32).
// Sparse Laplacian mat-vecs use L2-resident float atomics (8MB << 192MB L2).
// ---------------------------------------------------------------------------

typedef _Float16 half_t;
typedef _Float16 v16h __attribute__((ext_vector_type(16)));
typedef float    v8f  __attribute__((ext_vector_type(8)));

#define NTEAM  100000
#define NEDGE  3200000
#define NTILE  (NTEAM / 16)   // 6250 exact
#define ETILE  (NEDGE / 16)   // 200000 exact

// ---- workspace layout (float offsets). Needs ~58.5 MB. --------------------
static constexpr size_t N_PAD    = 100096;                 // 100000 -> mult of 256
static constexpr size_t OFF_DEG  = 0;                      // int[N_PAD]
static constexpr size_t OFF_DINV = N_PAD;                  // float[N_PAD]
static constexpr size_t OFF_W    = 2 * N_PAD;              // float[NEDGE]
static constexpr size_t TX_SZ    = 2000128;                // 100000*20 padded
static constexpr size_t OFF_TX1  = OFF_W + NEDGE;
static constexpr size_t OFF_TX2  = OFF_TX1 + TX_SZ;
static constexpr size_t OFF_TX3  = OFF_TX2 + TX_SZ;
static constexpr size_t OFF_TX4  = OFF_TX3 + TX_SZ;
static constexpr size_t OFF_HA   = OFF_TX4 + TX_SZ;        // float[100000*16]
static constexpr size_t OFF_HB   = OFF_HA + (size_t)NTEAM * 16;
static constexpr size_t OFF_BIAS = OFF_HB + (size_t)NTEAM * 16;  // bz48 bh48 bd96
static constexpr size_t OFF_HALF = OFF_BIAS + 256;         // half region (32B aligned)
// halves: Bconv [3][2][5][32][16] = 15360, Bdense [6][32][16] = 3072

// ---------------------------------------------------------------------------
__global__ void k_zero_f32(float* p, int n) {
  int i = blockIdx.x * blockDim.x + threadIdx.x;
  if (i < n) p[i] = 0.f;
}
__global__ void k_zero_i32(int* p, int n) {
  int i = blockIdx.x * blockDim.x + threadIdx.x;
  if (i < n) p[i] = 0;
}
__global__ void k_negcopy(float* __restrict__ o, const float* __restrict__ a, int n) {
  int i = blockIdx.x * blockDim.x + threadIdx.x;
  if (i < n) o[i] = -a[i];
}
__global__ void k_deg(const int* __restrict__ ei, int* __restrict__ deg) {
  int e = blockIdx.x * blockDim.x + threadIdx.x;
  if (e < NEDGE) atomicAdd(&deg[ei[e]], 1);
}
__global__ void k_dinv(const int* __restrict__ deg, float* __restrict__ dinv) {
  int i = blockIdx.x * blockDim.x + threadIdx.x;
  if (i < NTEAM) {
    int d = deg[i];
    dinv[i] = (d > 0) ? rsqrtf((float)d) : 0.f;
  }
}
__global__ void k_w(const int* __restrict__ ei, const float* __restrict__ dinv,
                    float* __restrict__ w) {
  int e = blockIdx.x * blockDim.x + threadIdx.x;
  if (e < NEDGE) w[e] = -dinv[ei[e]] * dinv[ei[NEDGE + e]];
}

// out[dst] += scale * w[e] * v[src]; one thread per (edge, 4-float chunk)
template <int CI>
__global__ void k_scatter(float* __restrict__ out, const float* __restrict__ v,
                          const int* __restrict__ ei, const float* __restrict__ w,
                          float scale) {
  constexpr int CH = CI / 4;
  int gid = blockIdx.x * blockDim.x + threadIdx.x;
  if (gid >= NEDGE * CH) return;
  int e  = gid / CH;
  int c0 = (gid - e * CH) * 4;
  int s = ei[e], d = ei[NEDGE + e];
  float ww = scale * w[e];
  float4 vv = *(const float4*)(v + (size_t)s * CI + c0);
  float* op = out + (size_t)d * CI + c0;
  atomicAdd(op + 0, ww * vv.x);
  atomicAdd(op + 1, ww * vv.y);
  atomicAdd(op + 2, ww * vv.z);
  atomicAdd(op + 3, ww * vv.w);
}

// ---- weight packing into WMMA B-layout (f16 32x16: lane=col, K-half=lane/16)
struct PackArgs {
  const float* Wxz[3]; const float* Wxh[3];
  const float* bxz[3]; const float* bhz[3]; const float* bxh[3]; const float* bhh[3];
  const float* Wd[6];  const float* bdn[6];
};

__global__ void k_pack(PackArgs pa, half_t* __restrict__ Bconv,
                       half_t* __restrict__ Bdense, float* __restrict__ bz,
                       float* __restrict__ bh, float* __restrict__ bdp) {
  int id = blockIdx.x * blockDim.x + threadIdx.x;
  if (id < 15360) {                              // conv blocks: (l*2+g)*5+k
    int blk = id >> 9, i = id & 511;
    int l = blk / 10, g = (blk / 5) % 2, k = blk % 5;
    int lane = i >> 4, el = i & 15;
    int K = ((el < 8) ? 0 : 16) + (lane >> 4) * 8 + (el & 7), n = lane & 15;
    int ci = (l == 0) ? 20 : 16;
    const float* W = g ? pa.Wxh[l] : pa.Wxz[l];  // [5][ci][16]
    float v = (K < ci) ? W[(k * ci + K) * 16 + n] : 0.f;
    Bconv[id] = (half_t)v;
  } else if (id < 18432) {                       // dense blocks
    int id2 = id - 15360;
    int blk = id2 >> 9, i = id2 & 511;
    int lane = i >> 4, el = i & 15;
    int K = ((el < 8) ? 0 : 16) + (lane >> 4) * 8 + (el & 7), n = lane & 15;
    int ad = (blk == 0) ? 32 : 8, bd2 = (blk == 5) ? 3 : 8;
    float v = (K < ad && n < bd2) ? pa.Wd[blk][K * bd2 + n] : 0.f;
    Bdense[id2] = (half_t)v;
  } else if (id < 18480) {                       // bz = b_xz + b_hz
    int j = id - 18432; int l = j / 16, c = j % 16;
    bz[j] = pa.bxz[l][c] + pa.bhz[l][c];
  } else if (id < 18528) {                       // bh = b_xh + b_hh
    int j = id - 18480; int l = j / 16, c = j % 16;
    bh[j] = pa.bxh[l][c] + pa.bhh[l][c];
  } else if (id < 18624) {                       // dense bias padded to 16
    int j = id - 18528; int L = j / 16, c = j % 16;
    int bd2 = (L == 5) ? 3 : 8;
    bdp[j] = (c < bd2) ? pa.bdn[L][c] : 0.f;
  }
}

// ---- gate kernel: per 16-node tile, 10 WMMAs (5 Chebyshev k x {z, h}) -----
template <int CI>
__global__ void k_gates(float* __restrict__ hout,
                        const float* __restrict__ tx0, const float* __restrict__ tx1,
                        const float* __restrict__ tx2, const float* __restrict__ tx3,
                        const float* __restrict__ tx4,
                        const half_t* __restrict__ Bz, const half_t* __restrict__ Bh,
                        const float* __restrict__ bz, const float* __restrict__ bh) {
  int lane = threadIdx.x & 31;
  int wv   = threadIdx.x >> 5;
  int tile = blockIdx.x * (blockDim.x >> 5) + wv;
  if (tile >= NTILE) return;                 // whole-wave exit: EXEC stays all-1
  int r = lane & 15, hi = lane >> 4;
  int row = tile * 16 + r;

  float bzv = bz[r], bhv = bh[r];            // C/D layout: N = lane%16
  v8f cz, ch;
#pragma unroll
  for (int j = 0; j < 8; j++) { cz[j] = bzv; ch[j] = bhv; }

  const float* txp[5] = {tx0, tx1, tx2, tx3, tx4};
#pragma unroll
  for (int k = 0; k < 5; k++) {
    const float* T = txp[k] + (size_t)row * CI;
    v16h a;
#pragma unroll
    for (int e = 0; e < 16; e++) {           // A layout: K = base + 8*hi + e%8
      int K = ((e < 8) ? 0 : 16) + hi * 8 + (e & 7);
      a[e] = (K < CI) ? (half_t)T[K] : (half_t)0.f;
    }
    v16h bm0 = *(const v16h*)(Bz + k * 512 + lane * 16);
    v16h bm1 = *(const v16h*)(Bh + k * 512 + lane * 16);
    cz = __builtin_amdgcn_wmma_f32_16x16x32_f16(false, a, false, bm0, (short)0, cz, false, false);
    ch = __builtin_amdgcn_wmma_f32_16x16x32_f16(false, a, false, bm1, (short)0, ch, false, false);
  }
#pragma unroll
  for (int j = 0; j < 8; j++) {
    int M = j + 8 * hi;
    float z  = 1.f / (1.f + __expf(-cz[j]));
    float ht = tanhf(ch[j]);
    float hv = (1.f - z) * ht;               // H = (1-Z)*Ht, then relu
    hout[(size_t)(tile * 16 + M) * 16 + r] = (hv > 0.f) ? hv : 0.f;
  }
}

// ---- edge MLP: one wave per 16 edges, 6 chained WMMAs, LDS re-layout ------
__global__ void __launch_bounds__(256)
k_mlp(const float* __restrict__ h, const int* __restrict__ ei,
      const half_t* __restrict__ Bd, const float* __restrict__ bd,
      float* __restrict__ out) {
  __shared__ float stage[8][16][17];
  int lane = threadIdx.x & 31;
  int wv   = threadIdx.x >> 5;
  int tile = blockIdx.x * 8 + wv;            // 25000 blocks exactly
  int r = lane & 15, hi = lane >> 4;
  int e = tile * 16 + r;
  int s = ei[e], d = ei[NEDGE + e];

  v16h a;
  {  // A = [h_src | h_dst] (K=32): e<8 -> src[8*hi+e], e>=8 -> dst[8*hi+e-8]
    const float4* hs = (const float4*)(h + (size_t)s * 16 + hi * 8);
    const float4* hd = (const float4*)(h + (size_t)d * 16 + hi * 8);
    float4 p0 = hs[0], p1 = hs[1], q0 = hd[0], q1 = hd[1];
    a[0]=(half_t)p0.x; a[1]=(half_t)p0.y; a[2]=(half_t)p0.z; a[3]=(half_t)p0.w;
    a[4]=(half_t)p1.x; a[5]=(half_t)p1.y; a[6]=(half_t)p1.z; a[7]=(half_t)p1.w;
    a[8]=(half_t)q0.x; a[9]=(half_t)q0.y; a[10]=(half_t)q0.z; a[11]=(half_t)q0.w;
    a[12]=(half_t)q1.x; a[13]=(half_t)q1.y; a[14]=(half_t)q1.z; a[15]=(half_t)q1.w;
  }

#pragma unroll
  for (int L = 0; L < 6; L++) {
    float bv = bd[L * 16 + r];
    v8f c;
#pragma unroll
    for (int j = 0; j < 8; j++) c[j] = bv;
    v16h b = *(const v16h*)(Bd + L * 512 + lane * 16);
    c = __builtin_amdgcn_wmma_f32_16x16x32_f16(false, a, false, b, (short)0, c, false, false);
    __syncthreads();
#pragma unroll
    for (int j = 0; j < 8; j++) {            // relu + stage at [M][N]
      float v = c[j];
      stage[wv][j + 8 * hi][r] = (v > 0.f) ? v : 0.f;
    }
    __syncthreads();
    if (L < 5) {                             // rebuild A (K=8 live, rest 0)
#pragma unroll
      for (int q = 0; q < 16; q++) a[q] = (half_t)0.f;
      if (hi == 0) {
#pragma unroll
        for (int q = 0; q < 8; q++) a[q] = (half_t)stage[wv][r][q];
      }
    }
  }
  if (hi == 0) {                             // softmax over 3 logits
    float v0 = stage[wv][r][0], v1 = stage[wv][r][1], v2 = stage[wv][r][2];
    float mx = fmaxf(v0, fmaxf(v1, v2));
    float e0 = __expf(v0 - mx), e1 = __expf(v1 - mx), e2 = __expf(v2 - mx);
    float inv = 1.f / (e0 + e1 + e2);
    out[(size_t)e * 3 + 0] = e0 * inv;
    out[(size_t)e * 3 + 1] = e1 * inv;
    out[(size_t)e * 3 + 2] = e2 * inv;
  }
}

// ---------------------------------------------------------------------------
template <int CI>
static void run_layer(const float* tx0, float* tx1, float* tx2, float* tx3,
                      float* tx4, float* hout, const half_t* Bz, const half_t* Bh,
                      const float* bz, const float* bh, const int* ei,
                      const float* w, hipStream_t stream) {
  const int n = NTEAM * CI;
  const int zb = (n + 255) / 256;
  const int sb = (NEDGE * (CI / 4) + 255) / 256;
  k_zero_f32<<<zb, 256, 0, stream>>>(tx1, n);
  k_scatter<CI><<<sb, 256, 0, stream>>>(tx1, tx0, ei, w, 1.f);      // T1 = L T0
  k_negcopy<<<zb, 256, 0, stream>>>(tx2, tx0, n);
  k_scatter<CI><<<sb, 256, 0, stream>>>(tx2, tx1, ei, w, 2.f);      // T2 = 2L T1 - T0
  k_negcopy<<<zb, 256, 0, stream>>>(tx3, tx1, n);
  k_scatter<CI><<<sb, 256, 0, stream>>>(tx3, tx2, ei, w, 2.f);      // T3
  k_negcopy<<<zb, 256, 0, stream>>>(tx4, tx2, n);
  k_scatter<CI><<<sb, 256, 0, stream>>>(tx4, tx3, ei, w, 2.f);      // T4
  k_gates<CI><<<(NTILE + 3) / 4, 128, 0, stream>>>(hout, tx0, tx1, tx2, tx3,
                                                   tx4, Bz, Bh, bz, bh);
}

extern "C" void kernel_launch(void* const* d_in, const int* in_sizes, int n_in,
                              void* d_out, int out_size, void* d_ws, size_t ws_size,
                              hipStream_t stream) {
  (void)in_sizes; (void)n_in; (void)out_size; (void)ws_size;
  // jax tree-flatten order (dict keys sorted at every level):
  //  [0] edge_index; [1..12] dense W0,b0..W5,b5; [13] embed;
  //  per layer l at base 14+12*l: W_hh,W_hr,W_hz,W_xh,W_xr,W_xz,
  //                               b_hh,b_hr,b_hz,b_xh,b_xr,b_xz
  const int*   ei    = (const int*)d_in[0];
  const float* embed = (const float*)d_in[13];
  PackArgs pa;
  for (int d = 0; d < 6; d++) {
    pa.Wd[d]  = (const float*)d_in[1 + 2 * d];
    pa.bdn[d] = (const float*)d_in[2 + 2 * d];
  }
  for (int l = 0; l < 3; l++) {
    int b = 14 + 12 * l;
    pa.Wxh[l] = (const float*)d_in[b + 3];
    pa.Wxz[l] = (const float*)d_in[b + 5];
    pa.bhh[l] = (const float*)d_in[b + 6];
    pa.bhz[l] = (const float*)d_in[b + 8];
    pa.bxh[l] = (const float*)d_in[b + 9];
    pa.bxz[l] = (const float*)d_in[b + 11];
  }

  float*  ws   = (float*)d_ws;
  int*    deg  = (int*)(ws + OFF_DEG);
  float*  dinv = ws + OFF_DINV;
  float*  w    = ws + OFF_W;
  float*  tx1  = ws + OFF_TX1;
  float*  tx2  = ws + OFF_TX2;
  float*  tx3  = ws + OFF_TX3;
  float*  tx4  = ws + OFF_TX4;
  float*  hA   = ws + OFF_HA;
  float*  hB   = ws + OFF_HB;
  float*  bz   = ws + OFF_BIAS;
  float*  bh   = bz + 48;
  float*  bdp  = bh + 48;
  half_t* Bconv  = (half_t*)(ws + OFF_HALF);
  half_t* Bdense = Bconv + 15360;

  const int EB = (NEDGE + 255) / 256;   // 12500
  const int NB = (NTEAM + 255) / 256;   // 391

  k_zero_i32<<<NB, 256, 0, stream>>>(deg, NTEAM);
  k_deg<<<EB, 256, 0, stream>>>(ei, deg);
  k_dinv<<<NB, 256, 0, stream>>>(deg, dinv);
  k_w<<<EB, 256, 0, stream>>>(ei, dinv, w);
  k_pack<<<73, 256, 0, stream>>>(pa, Bconv, Bdense, bz, bh, bdp);

  // layer 0: ci=20, in=embed, out=hA;  layer 1: hA->hB;  layer 2: hB->hA
  run_layer<20>(embed, tx1, tx2, tx3, tx4, hA,
                Bconv + 0 * 5120, Bconv + 0 * 5120 + 2560, bz + 0, bh + 0,
                ei, w, stream);
  run_layer<16>(hA, tx1, tx2, tx3, tx4, hB,
                Bconv + 1 * 5120, Bconv + 1 * 5120 + 2560, bz + 16, bh + 16,
                ei, w, stream);
  run_layer<16>(hB, tx1, tx2, tx3, tx4, hA,
                Bconv + 2 * 5120, Bconv + 2 * 5120 + 2560, bz + 32, bh + 32,
                ei, w, stream);

  k_mlp<<<NEDGE / 128, 256, 0, stream>>>(hA, ei, Bdense, bdp, (float*)d_out);
}
